// SwinAttention_26663156973982
// MI455X (gfx1250) — matrix operand order repaired and checked by reading, not compiled, not run on hardware
//
#include <hip/hip_runtime.h>

// ---------------------------------------------------------------------------
// CDNA5 (gfx1250, wave32) Swin window-attention: bf16 WMMA pipeline.
//   v_wmma_f32_16x16x32_bf16 for QKV gemm, QK^T, PV, and output projection.
// ---------------------------------------------------------------------------

typedef __bf16 bf16_t;
typedef __bf16 v16bf __attribute__((ext_vector_type(16)));
typedef float  v8f   __attribute__((ext_vector_type(8)));

union AB32 { uint4 q[2]; v16bf v; };
union BFU  { unsigned short u; bf16_t b; };

__device__ __forceinline__ bf16_t f2bf(float f) {
  unsigned u = __float_as_uint(f);
  u += 0x7FFFu + ((u >> 16) & 1u);          // round-to-nearest-even
  BFU x; x.u = (unsigned short)(u >> 16);
  return x.b;
}

__device__ __forceinline__ v16bf frag_zero() {
  AB32 z; z.q[0] = make_uint4(0u,0u,0u,0u); z.q[1] = make_uint4(0u,0u,0u,0u);
  return z.v;
}

// A-matrix 16x32 bf16 fragment (contiguous k): lane -> row = lane&15,
// k0 = (lane>>4)*16; holds K[k0..k0+15].
__device__ __forceinline__ v16bf frag_a(const bf16_t* tile, int ld, int lane) {
  const int r  = lane & 15;
  const int k0 = (lane >> 4) * 16;
  AB32 u;
  u.q[0] = *(const uint4*)(tile + (size_t)r * ld + k0);
  u.q[1] = *(const uint4*)(tile + (size_t)r * ld + k0 + 8);
  return u.v;
}

// B-matrix 32x16 bf16 fragment payload: 16 contiguous bf16 (lane = column,
// K range (lane>>4)*16 .. +15) -- caller supplies the per-lane pointer.
__device__ __forceinline__ v16bf frag_row16(const bf16_t* p) {
  AB32 u; const uint4* q = (const uint4*)p;
  u.q[0] = q[0]; u.q[1] = q[1];
  return u.v;
}

#define WMMA_BF16(A,B,C) \
  __builtin_amdgcn_wmma_f32_16x16x32_bf16(false,(A),false,(B),(short)0,(C),false,false)

template<int M>
__device__ __forceinline__ float swz_xor(float v) {  // xor-swizzle within 32 lanes
  return __int_as_float(__builtin_amdgcn_ds_swizzle(__float_as_int(v), (M << 10) | 0x1f));
}

// ------------------------------ f32 -> bf16 --------------------------------
struct bf16x4 { bf16_t v[4]; };

__global__ __launch_bounds__(256) void swin_cvt_bf16(const float4* __restrict__ in,
                                                     bf16x4* __restrict__ out, int n4) {
  const int i = blockIdx.x * 256 + threadIdx.x;
  if (i < n4) {
    float4 f = in[i];
    bf16x4 o;
    o.v[0] = f2bf(f.x); o.v[1] = f2bf(f.y); o.v[2] = f2bf(f.z); o.v[3] = f2bf(f.w);
    out[i] = o;
  }
}

// ------------------------- QKV GEMM + scatter ------------------------------
// X[100352,512](bf16) x W[1536,512](bf16, row=out-feature) + b -> Q/K/Vt
// Q,K : [B*H][64(pad)][32] bf16 (Q pre-scaled by 1/sqrt(32))
// Vt  : [B*H][32][64(pad)] bf16 (transposed so PV B-fragments are contiguous)
__global__ __launch_bounds__(128) void swin_qkv_gemm(
    const bf16_t* __restrict__ X, const bf16_t* __restrict__ W,
    const float* __restrict__ bqkv,
    bf16_t* __restrict__ Q, bf16_t* __restrict__ K, bf16_t* __restrict__ Vt)
{
  const int lane = threadIdx.x & 31;
  const int wave = threadIdx.x >> 5;
  const int m0 = blockIdx.x * 128 + wave * 32;
  const int n0 = blockIdx.y * 64;

  v8f acc[2][4] = {};
  const bf16_t* Ar0 = X + (size_t)m0 * 512;
  const bf16_t* Ar1 = X + (size_t)(m0 + 16) * 512;

  for (int kk = 0; kk < 512; kk += 32) {
    __builtin_prefetch(Ar0 + kk + 64, 0, 1);
    __builtin_prefetch(Ar1 + kk + 64, 0, 1);
    v16bf a0 = frag_a(Ar0 + kk, 512, lane);
    v16bf a1 = frag_a(Ar1 + kk, 512, lane);
#pragma unroll
    for (int t = 0; t < 4; ++t) {
      v16bf bt = frag_row16(W + (size_t)(n0 + t*16 + (lane & 15)) * 512 + kk + ((lane >> 4) * 16));
      acc[0][t] = WMMA_BF16(a0, bt, acc[0][t]);
      acc[1][t] = WMMA_BF16(a1, bt, acc[1][t]);
    }
  }

  const int half = lane >> 4, nl = lane & 15;
#pragma unroll
  for (int t = 0; t < 4; ++t) {
    const int n = n0 + t*16 + nl;
    const float bias = bqkv[n];
    const int mat = n >> 9;          // 0=q, 1=k, 2=v (uniform per tile)
    const int c = n & 511;
    const int h = c >> 5, d = c & 31;
#pragma unroll
    for (int i = 0; i < 2; ++i) {
#pragma unroll
      for (int r = 0; r < 8; ++r) {
        const int m = m0 + i*16 + r + half*8;   // C-tile: VGPR r <-> rows r / r+8
        const int bwin = m / 49;
        const int tt = m - bwin * 49;
        const size_t bh = (size_t)bwin * 16 + h;
        const float val = acc[i][t][r] + bias;
        if (mat == 0)      Q [bh*2048 + (size_t)tt*32 + d] = f2bf(val * 0.1767766952966369f);
        else if (mat == 1) K [bh*2048 + (size_t)tt*32 + d] = f2bf(val);
        else               Vt[bh*2048 + (size_t)d*64 + tt] = f2bf(val);
      }
    }
  }
}

// ---------------------------- attention core -------------------------------
// One workgroup (4 waves) per (b,h). Wave w owns S/O rows [16w,16w+16).
__global__ __launch_bounds__(128) void swin_attn(
    const bf16_t* __restrict__ Q, const bf16_t* __restrict__ K,
    const bf16_t* __restrict__ Vt,
    const float* __restrict__ amask, const float* __restrict__ relt,
    bf16_t* __restrict__ AO)
{
  __shared__ float s_bias[169 * 16];            // rel-pos table
  __shared__ float s_mask[49 * 49];             // this window's mask
  __shared__ __align__(16) bf16_t s_p[4][16 * 64];  // per-wave P staging

  const int tid  = threadIdx.x;
  const int lane = tid & 31, wave = tid >> 5;
  const int half = lane >> 4, jloc = lane & 15;
  const int bh = blockIdx.x;
  const int b = bh >> 4, h = bh & 15;

  for (int i = tid; i < 169 * 16; i += 128) s_bias[i] = relt[i];
  for (int i = tid; i < 49 * 49; i += 128) s_mask[i] = amask[(size_t)b * 2401 + i];
  __syncthreads();

  const bf16_t* Qb = Q  + (size_t)bh * 2048;
  const bf16_t* Kb = K  + (size_t)bh * 2048;
  const bf16_t* Vb = Vt + (size_t)bh * 2048;

  // S = (Q*scale) @ K^T, four 16x16 tiles along j, single K=32 step.
  v16bf qa;
  {
    const int qrow = wave * 16 + jloc;
    qa = (qrow < 49) ? frag_a(Qb + (size_t)(wave * 16) * 32, 32, lane) : frag_zero();
  }
  v8f s[4] = {};
#pragma unroll
  for (int t = 0; t < 4; ++t) {
    const int col = t * 16 + jloc;
    v16bf kb = (col < 49) ? frag_row16(Kb + (size_t)col * 32 + half * 16) : frag_zero();
    s[t] = WMMA_BF16(qa, kb, s[t]);
  }

  // + rel-pos bias + mask, then row softmax (halves reduce rows r and r+8).
  float pv[4][8];
#pragma unroll
  for (int r = 0; r < 8; ++r) {
    const int i = wave * 16 + r + half * 8;
    const bool iok = i < 49;
    const int ih = i / 7, iw = i - ih * 7;
    float mx = -1e30f;
#pragma unroll
    for (int t = 0; t < 4; ++t) {
      const int j = t * 16 + jloc;
      float v = -1e30f;
      if (iok && j < 49) {
        const int jh = j / 7, jw = j - jh * 7;
        const int idx = (ih - jh + 6) * 13 + (iw - jw + 6);
        v = s[t][r] + s_bias[idx * 16 + h] + s_mask[i * 49 + j];
      }
      pv[t][r] = v;
      mx = fmaxf(mx, v);
    }
    mx = fmaxf(mx, swz_xor<1>(mx));
    mx = fmaxf(mx, swz_xor<2>(mx));
    mx = fmaxf(mx, swz_xor<4>(mx));
    mx = fmaxf(mx, swz_xor<8>(mx));
    float sum = 0.f;
#pragma unroll
    for (int t = 0; t < 4; ++t) {
      const float e = __expf(pv[t][r] - mx);
      pv[t][r] = e;
      sum += e;
    }
    sum += swz_xor<1>(sum);
    sum += swz_xor<2>(sum);
    sum += swz_xor<4>(sum);
    sum += swz_xor<8>(sum);
    const float inv = 1.0f / sum;
#pragma unroll
    for (int t = 0; t < 4; ++t) pv[t][r] *= inv;
  }

  // Stage P (C-layout regs -> row-major bf16 LDS tile [16 x 64]).
  bf16_t* myP = s_p[wave];
#pragma unroll
  for (int r = 0; r < 8; ++r) {
    const int il = r + half * 8;
#pragma unroll
    for (int t = 0; t < 4; ++t)
      myP[il * 64 + t * 16 + jloc] = f2bf(pv[t][r]);
  }

  // O = P @ V : A frags from LDS, B frags contiguous from Vt.
  v8f o[2] = {};
#pragma unroll
  for (int kh = 0; kh < 2; ++kh) {
    v16bf pa = frag_a(myP + kh * 32, 64, lane);
#pragma unroll
    for (int n = 0; n < 2; ++n) {
      v16bf vb = frag_row16(Vb + (size_t)(n * 16 + jloc) * 64 + kh * 32 + half * 16);
      o[n] = WMMA_BF16(pa, vb, o[n]);
    }
  }

#pragma unroll
  for (int n = 0; n < 2; ++n) {
    const int d = n * 16 + jloc;
#pragma unroll
    for (int r = 0; r < 8; ++r) {
      const int i = wave * 16 + r + half * 8;
      if (i < 49)
        AO[((size_t)b * 49 + i) * 512 + h * 32 + d] = f2bf(o[n][r]);
    }
  }
}

// -------------------------- output projection ------------------------------
__global__ __launch_bounds__(128) void swin_proj_gemm(
    const bf16_t* __restrict__ A, const bf16_t* __restrict__ W,
    const float* __restrict__ bo, float* __restrict__ out)
{
  const int lane = threadIdx.x & 31;
  const int wave = threadIdx.x >> 5;
  const int m0 = blockIdx.x * 128 + wave * 32;
  const int n0 = blockIdx.y * 64;

  v8f acc[2][4] = {};
  const bf16_t* Ar0 = A + (size_t)m0 * 512;
  const bf16_t* Ar1 = A + (size_t)(m0 + 16) * 512;

  for (int kk = 0; kk < 512; kk += 32) {
    __builtin_prefetch(Ar0 + kk + 64, 0, 1);
    __builtin_prefetch(Ar1 + kk + 64, 0, 1);
    v16bf a0 = frag_a(Ar0 + kk, 512, lane);
    v16bf a1 = frag_a(Ar1 + kk, 512, lane);
#pragma unroll
    for (int t = 0; t < 4; ++t) {
      v16bf bt = frag_row16(W + (size_t)(n0 + t*16 + (lane & 15)) * 512 + kk + ((lane >> 4) * 16));
      acc[0][t] = WMMA_BF16(a0, bt, acc[0][t]);
      acc[1][t] = WMMA_BF16(a1, bt, acc[1][t]);
    }
  }

  const int half = lane >> 4, nl = lane & 15;
#pragma unroll
  for (int t = 0; t < 4; ++t) {
    const int n = n0 + t*16 + nl;
    const float bias = bo[n];
#pragma unroll
    for (int i = 0; i < 2; ++i)
#pragma unroll
      for (int r = 0; r < 8; ++r) {
        const int m = m0 + i*16 + r + half*8;
        out[(size_t)m * 512 + n] = acc[i][t][r] + bias;
      }
  }
}

// ------------------------------- launcher ----------------------------------
extern "C" void kernel_launch(void* const* d_in, const int* in_sizes, int n_in,
                              void* d_out, int out_size, void* d_ws, size_t ws_size,
                              hipStream_t stream) {
  (void)in_sizes; (void)n_in; (void)out_size; (void)ws_size;

  const float* hidden = (const float*)d_in[0];   // [2048,49,512]
  const float* amask  = (const float*)d_in[1];   // [2048,49,49]
  const float* qkvw   = (const float*)d_in[2];   // [1536,512]
  const float* qkvb   = (const float*)d_in[3];   // [1536]
  const float* outw   = (const float*)d_in[4];   // [512,512]
  const float* outb   = (const float*)d_in[5];   // [512]
  const float* relt   = (const float*)d_in[6];   // [169,16]
  float* out = (float*)d_out;                    // [2048,49,512]

  char* p = (char*)d_ws;
  bf16_t* Xbf  = (bf16_t*)p; p += (size_t)100352 * 512 * 2;   // 102.8 MB
  bf16_t* Wqkv = (bf16_t*)p; p += (size_t)1536 * 512 * 2;
  bf16_t* Wout = (bf16_t*)p; p += (size_t)512 * 512 * 2;
  bf16_t* Qb   = (bf16_t*)p; p += (size_t)32768 * 2048 * 2;   // 134 MB
  bf16_t* Kb   = (bf16_t*)p; p += (size_t)32768 * 2048 * 2;
  bf16_t* Vt   = (bf16_t*)p; p += (size_t)32768 * 2048 * 2;
  bf16_t* AO   = Xbf;  // Xbf is dead after the QKV gemm; reuse it for attn out

  {
    const int n4 = (100352 * 512) / 4;
    swin_cvt_bf16<<<dim3((n4 + 255) / 256), 256, 0, stream>>>((const float4*)hidden, (bf16x4*)Xbf, n4);
  }
  {
    const int n4 = (1536 * 512) / 4;
    swin_cvt_bf16<<<dim3((n4 + 255) / 256), 256, 0, stream>>>((const float4*)qkvw, (bf16x4*)Wqkv, n4);
  }
  {
    const int n4 = (512 * 512) / 4;
    swin_cvt_bf16<<<dim3((n4 + 255) / 256), 256, 0, stream>>>((const float4*)outw, (bf16x4*)Wout, n4);
  }

  swin_qkv_gemm<<<dim3(784, 24), 128, 0, stream>>>(Xbf, Wqkv, qkvb, Qb, Kb, Vt);
  swin_attn    <<<dim3(32768),   128, 0, stream>>>(Qb, Kb, Vt, amask, relt, AO);
  swin_proj_gemm<<<dim3(784, 8), 128, 0, stream>>>(AO, Wout, outb, out);
}
